// SpatialAttention_34308198761263
// MI455X (gfx1250) — matrix operand order repaired and checked by reading, not compiled, not run on hardware
//
#include <hip/hip_runtime.h>

// MI455X / gfx1250, wave32. Spatial attention:
//   q,k,v = x @ W^T + b  (f16 WMMA, f32 acc)
//   att   = softmax((q k^T * 1/8) * attention_weights)   per (t,h)
//   out   = att @ v, written in (H,N,T,D)-contiguous order
//
// N=256 tokens, T=128 steps, D=64, H=8.

typedef __attribute__((ext_vector_type(16))) _Float16 v16h;
typedef __attribute__((ext_vector_type(8)))  float    v8f;
typedef __attribute__((ext_vector_type(4)))  unsigned int v4u;
typedef __attribute__((ext_vector_type(8)))  int      v8i;
typedef __attribute__((ext_vector_type(4)))  int      v4i;

#define WMMA_F16(a, b, c) \
  __builtin_amdgcn_wmma_f32_16x16x32_f16(false, (a), false, (b), (short)0, (c), false, false)

union H16 { v16h h; uint4 q[2]; };

// Load a 16-half fragment as two 16-byte chunks (global or LDS; compiler picks
// global_load_b128 / ds_load_b128 from the address space).
__device__ inline v16h ldfrag(const _Float16* base, int off0, int off1) {
  H16 u;
  u.q[0] = *(const uint4*)(base + off0);
  u.q[1] = *(const uint4*)(base + off1);
  return u.h;
}

// ---------------------------------------------------------------------------
// Kernel 0: f32 -> f16 conversion (x and the three weight matrices)
// ---------------------------------------------------------------------------
__global__ void cvt_f32_to_f16(const float* __restrict__ in,
                               _Float16* __restrict__ out, int n) {
  int i = blockIdx.x * blockDim.x + threadIdx.x;
  int stride = gridDim.x * blockDim.x;
  for (; i < n; i += stride) out[i] = (_Float16)in[i];
}

// ---------------------------------------------------------------------------
// Kernel 1: projections. grid = (NT/128, 3). Block = 256 threads = 8 waves.
// Wave w computes 16 rows (r0..r0+15) x 512 cols, in 4 chunks of 128 cols.
// Each chunk's W panel (128 rows x 64 halves) is staged once in LDS with a
// 72-half (144B) row pitch: 36-dword stride -> the 16 lanes of a b128
// fragment read hit 16 distinct banks (conflict-free).
// Outputs (f16):
//   mat 0 -> Qg[(t*8+h)*256 + n][d]   (t,h,n,d)
//   mat 1 -> Kg[(t*8+h)*256 + n][d]   (t,h,n,d)
//   mat 2 -> Vt[(t*8+h)*64 + d][n]    (t,h,d,n)  (transposed for PV B-frags)
// ---------------------------------------------------------------------------
__global__ __launch_bounds__(256) void proj_kernel(
    const _Float16* __restrict__ x16, const _Float16* __restrict__ W16all,
    const float* __restrict__ bq, const float* __restrict__ bk,
    const float* __restrict__ bv, _Float16* __restrict__ Qg,
    _Float16* __restrict__ Kg, _Float16* __restrict__ Vt) {
  __shared__ __align__(16) _Float16 Wsh[128 * 72];  // 18 KB padded W panel

  const int mat = blockIdx.y;
  const _Float16* W16 = W16all + mat * 512 * 64;
  const float* bias = (mat == 0) ? bq : (mat == 1) ? bk : bv;
  _Float16* dst = (mat == 0) ? Qg : (mat == 1) ? Kg : Vt;

  const int lane = threadIdx.x & 31;
  const int w    = threadIdx.x >> 5;
  const int l16  = lane & 15;
  const int g    = lane >> 4;
  const int r0   = blockIdx.x * 128 + w * 16;

  // A fragments for this wave's 16 rows (K = 0..31 and 32..63).
  const int rA = r0 + l16;
  const v16h a0 = ldfrag(x16, rA * 64 +      g * 8, rA * 64 + 16 + g * 8);
  const v16h a1 = ldfrag(x16, rA * 64 + 32 + g * 8, rA * 64 + 48 + g * 8);

  for (int chunk = 0; chunk < 4; ++chunk) {
    // Stage this chunk's 128 W rows into padded LDS (8 x uint4 per row).
    {
      const uint4* wsrc = (const uint4*)(W16 + chunk * 128 * 64);
      for (int i = threadIdx.x; i < 1024; i += 256) {
        const int r = i >> 3, cpt = i & 7;
        *(uint4*)(Wsh + r * 72 + cpt * 8) = wsrc[i];
      }
    }
    __syncthreads();

    v8f acc[8] = {};
#pragma unroll
    for (int j = 0; j < 8; ++j) {
      const int jr = j * 16 + l16;  // chunk-local output column (= W row)
      const v16h b0 = ldfrag(Wsh, jr * 72 +      g * 8, jr * 72 + 16 + g * 8);
      const v16h b1 = ldfrag(Wsh, jr * 72 + 32 + g * 8, jr * 72 + 48 + g * 8);
      acc[j] = WMMA_F16(a0, b0, acc[j]);
      acc[j] = WMMA_F16(a1, b1, acc[j]);
    }
    // bias + convert + scatter-store
#pragma unroll
    for (int j = 0; j < 8; ++j) {
      const int c  = chunk * 128 + j * 16 + l16;
      const float bval = bias[c];
      const int h = c >> 6, d = c & 63;
#pragma unroll
      for (int i = 0; i < 8; ++i) {
        const int r = r0 + i + 8 * g;
        const int n = r >> 7, t = r & 127;
        const _Float16 hv = (_Float16)(acc[j][i] + bval);
        if (mat == 2)
          dst[((t * 8 + h) * 64 + d) * 256 + n] = hv;
        else
          dst[((t * 8 + h) * 256 + n) * 64 + d] = hv;
      }
    }
    __syncthreads();  // protect Wsh before next chunk overwrites it
  }
}

// ---------------------------------------------------------------------------
// Kernel 2: attention. grid = T*H = 1024 blocks, one (t,h) each.
// 256 threads = 8 waves; wave w handles row-strips w and w+8 (16 rows each).
// K(t,h) is staged into LDS by the Tensor Data Mover (one tensor_load_to_lds
// issued by wave 0): a 1-D line of 16384 f16 elements with pad_interval =
// 32 dwords / pad_amount = 4 dwords, i.e. the TDM deposits K with a 144B row
// pitch (72 halves) -> bank-conflict-free b128 fragment reads.
// ---------------------------------------------------------------------------
__global__ __launch_bounds__(256) void attn_kernel(
    const _Float16* __restrict__ Qg, const _Float16* __restrict__ Kg,
    const _Float16* __restrict__ Vt, const float* __restrict__ aw,
    float* __restrict__ out) {
  __shared__ __align__(16) _Float16 Ksh[256 * 72];   // 36 KB padded K stage
  __shared__ __align__(16) _Float16 panel[8][512];   // 8 KB per-wave P relayout

  const int th = blockIdx.x;
  const int t = th >> 3, h = th & 7;
  const _Float16* Qb  = Qg + th * (256 * 64);
  const _Float16* Kb  = Kg + th * (256 * 64);
  const _Float16* Vb  = Vt + th * (64 * 256);
  const float*    awb = aw + (size_t)th * 65536;

  // --- TDM: stage K(t,h) into LDS ----------------------------------------
  if (threadIdx.x < 32) {
    const unsigned long long ga = (unsigned long long)(uintptr_t)Kb;
    const unsigned int lds_off = (unsigned int)(uintptr_t)(&Ksh[0]);
    // D# group 0: count=1 | lds_addr | global_addr[56:0] | type=2
    v4u g0;
    g0.x = 1u;                       // count=1, is_restore=0, gather off
    g0.y = lds_off;                  // LDS byte address
    g0.z = (unsigned int)(ga & 0xffffffffu);
    g0.w = (unsigned int)((ga >> 32) & 0x01ffffffu) | (2u << 30);  // type=2
    // D# group 1: data_size=2B, pad 4 dwords every 32 dwords, 1-D line of
    // 16384 elements (tensor_dim0 = tile_dim0 = 16384, tensor_dim1 = 1).
    v8i g1;
    g1[0] = (int)((1u << 16) |            // data_size = 2 bytes
                  (1u << 20) |            // pad_enable
                  (4u << 22) |            // pad_interval: 32 dwords
                  (3u << 25));            // pad_amount: 4 dwords (16B)
    g1[1] = (int)(0x4000u << 16);         // tensor_dim0[15:0] = 16384
    g1[2] = (int)(1u << 16);              // tensor_dim0[31:16]=0, tensor_dim1=1
    g1[3] = (int)(0x4000u << 16);         // tile_dim0 = 16384
    g1[4] = 0;                            // tile_dim1 = tile_dim2 = unused
    g1[5] = 16384;                        // tensor_dim0_stride
    g1[6] = 0;                            // stride hi, tensor_dim1_stride lo
    g1[7] = 0;
    // Benign higher-dim groups (dims = 1, tiles unused).
    v4i g2; g2[0] = 1; g2[1] = 1; g2[2] = 16384; g2[3] = 0;
    v4i g3; g3[0] = 16384; g3[1] = (int)(1u << 16); g3[2] = 0; g3[3] = 0;
    // This toolchain exposes the 6-arg form: extra int32x8 group, zero-filled.
    v8i g4 = {0, 0, 0, 0, 0, 0, 0, 0};
    __builtin_amdgcn_tensor_load_to_lds(g0, g1, g2, g3, g4, 0);
    __builtin_amdgcn_s_wait_tensorcnt(0);
  }
  __syncthreads();

  const int lane = threadIdx.x & 31;
  const int w    = threadIdx.x >> 5;
  const int l16  = lane & 15;
  const int g    = lane >> 4;
  _Float16* pan  = &panel[w][0];

  for (int s = 0; s < 2; ++s) {
    const int m0 = (w + s * 8) * 16;

    // Prefetch this strip's attention_weights rows (16 KB) into cache; this
    // is the dominant HBM stream (268 MB total).
    {
      const float* awstrip = awb + m0 * 256;
      for (int i = lane; i < 256; i += 32)
        __builtin_prefetch(awstrip + i * 16, 0, 1);
    }

    // --- S = Q K^T (16 x 256 strip) --------------------------------------
    const int rA = m0 + l16;
    const v16h qa0 = ldfrag(Qb, rA * 64 +      g * 8, rA * 64 + 16 + g * 8);
    const v16h qa1 = ldfrag(Qb, rA * 64 + 32 + g * 8, rA * 64 + 48 + g * 8);

    v8f acc[16] = {};
#pragma unroll
    for (int j = 0; j < 16; ++j) {
      const int mB = j * 16 + l16;  // K-matrix row == S column
      const v16h b0 = ldfrag(Ksh, mB * 72 +      g * 8, mB * 72 + 16 + g * 8);
      acc[j] = WMMA_F16(qa0, b0, acc[j]);
      const v16h b1 = ldfrag(Ksh, mB * 72 + 32 + g * 8, mB * 72 + 48 + g * 8);
      acc[j] = WMMA_F16(qa1, b1, acc[j]);
    }

    // --- scale * attention_weights ---------------------------------------
    const float scale = 0.125f;  // 1/sqrt(64)
#pragma unroll
    for (int j = 0; j < 16; ++j) {
#pragma unroll
      for (int i = 0; i < 8; ++i) {
        const float wv = awb[(m0 + i + 8 * g) * 256 + j * 16 + l16];
        acc[j][i] = acc[j][i] * scale * wv;
      }
    }

    // --- softmax over the 256 columns of each row -------------------------
    // Row m = i + 8g lives in lanes {16g .. 16g+15}; xor masks 1,2,4,8 stay
    // inside that half-group.
    float mx[8], rinv[8];
#pragma unroll
    for (int i = 0; i < 8; ++i) {
      float m = acc[0][i];
#pragma unroll
      for (int j = 1; j < 16; ++j) m = fmaxf(m, acc[j][i]);
      m = fmaxf(m, __shfl_xor(m, 1, 32));
      m = fmaxf(m, __shfl_xor(m, 2, 32));
      m = fmaxf(m, __shfl_xor(m, 4, 32));
      m = fmaxf(m, __shfl_xor(m, 8, 32));
      mx[i] = m;
    }
#pragma unroll
    for (int i = 0; i < 8; ++i) {
      float sacc = 0.0f;
#pragma unroll
      for (int j = 0; j < 16; ++j) {
        const float p = __expf(acc[j][i] - mx[i]);
        acc[j][i] = p;
        sacc += p;
      }
      sacc += __shfl_xor(sacc, 1, 32);
      sacc += __shfl_xor(sacc, 2, 32);
      sacc += __shfl_xor(sacc, 4, 32);
      sacc += __shfl_xor(sacc, 8, 32);
      rinv[i] = 1.0f / sacc;
    }

    // --- O = P V (16 x 64) ------------------------------------------------
    v8f o[4] = {};
#pragma unroll
    for (int kk = 0; kk < 8; ++kk) {
      // Re-lay 2 D-layout tiles (cols kk*32..kk*32+31) as a 16x32 A-panel.
#pragma unroll
      for (int jj = 0; jj < 2; ++jj) {
#pragma unroll
        for (int i = 0; i < 8; ++i) {
          pan[(i + 8 * g) * 32 + jj * 16 + l16] =
              (_Float16)(acc[2 * kk + jj][i] * rinv[i]);
        }
      }
      // LDS ops are in-order within a wave; wait + compiler fence anyway.
      asm volatile("s_wait_dscnt 0" ::: "memory");
      const v16h pa = ldfrag(pan, l16 * 32 + g * 8, l16 * 32 + 16 + g * 8);
#pragma unroll
      for (int j2 = 0; j2 < 4; ++j2) {
        const int d = j2 * 16 + l16;  // V^T row == output column
        const v16h vb = ldfrag(Vb, d * 256 + kk * 32 + g * 8,
                                   d * 256 + kk * 32 + 16 + g * 8);
        o[j2] = WMMA_F16(pa, vb, o[j2]);
      }
    }

    // --- store: out flat order is (H, N, T, D) ---------------------------
#pragma unroll
    for (int j2 = 0; j2 < 4; ++j2) {
      const int d = j2 * 16 + l16;
#pragma unroll
      for (int i = 0; i < 8; ++i) {
        const int n = m0 + i + 8 * g;
        out[(((size_t)h * 256 + n) * 128 + t) * 64 + d] = o[j2][i];
      }
    }
  }
}

// ---------------------------------------------------------------------------
// Host launcher
// ---------------------------------------------------------------------------
extern "C" void kernel_launch(void* const* d_in, const int* in_sizes, int n_in,
                              void* d_out, int out_size, void* d_ws,
                              size_t ws_size, hipStream_t stream) {
  (void)in_sizes; (void)n_in; (void)out_size; (void)ws_size;
  const float* x  = (const float*)d_in[0];
  const float* Wq = (const float*)d_in[1];
  const float* bq = (const float*)d_in[2];
  const float* Wk = (const float*)d_in[3];
  const float* bk = (const float*)d_in[4];
  const float* Wv = (const float*)d_in[5];
  const float* bv = (const float*)d_in[6];
  const float* aw = (const float*)d_in[7];
  float* out = (float*)d_out;

  // Workspace layout (f16 elements): x16 | W16(q,k,v) | Qg | Kg | Vt
  _Float16* wsh = (_Float16*)d_ws;
  _Float16* x16 = wsh;                         // 32768*64   = 2,097,152
  _Float16* W16 = x16 + 2097152;               // 3*512*64   =    98,304
  _Float16* Qg  = W16 + 98304;                 // 128*8*256*64 = 16,777,216
  _Float16* Kg  = Qg + 16777216;
  _Float16* Vt  = Kg + 16777216;

  cvt_f32_to_f16<<<1024, 256, 0, stream>>>(x,  x16,          2097152);
  cvt_f32_to_f16<<<128,  256, 0, stream>>>(Wq, W16 + 0,      32768);
  cvt_f32_to_f16<<<128,  256, 0, stream>>>(Wk, W16 + 32768,  32768);
  cvt_f32_to_f16<<<128,  256, 0, stream>>>(Wv, W16 + 65536,  32768);

  proj_kernel<<<dim3(256, 3), 256, 0, stream>>>(x16, W16, bq, bk, bv, Qg, Kg, Vt);
  attn_kernel<<<1024, 256, 0, stream>>>(Qg, Kg, Vt, aw, out);
}
